// CommNetMLP_62912680952056
// MI455X (gfx1250) — matrix operand order repaired and checked by reading, not compiled, not run on hardware
//
#include <hip/hip_runtime.h>

#define Bb   64
#define Nn   100
#define Hh   128
#define MT   7        // 7 m-tiles of 16 -> 112 padded rows
#define MPAD 112
#define Pp   2
#define Aa   5

typedef __attribute__((ext_vector_type(16))) __bf16 v16bf;
typedef __attribute__((ext_vector_type(8)))  float  v8f;

__device__ inline unsigned short f2bfbits(float f) {
    union { float f; unsigned u; } v; v.f = f;
    unsigned r = v.u + 0x7FFFu + ((v.u >> 16) & 1u);   // round-to-nearest-even
    return (unsigned short)(r >> 16);
}
__device__ inline float bf2f(unsigned short h) {
    union { unsigned u; float f; } v; v.u = ((unsigned)h) << 16;
    return v.f;
}
__device__ inline v16bf pack16(uint4 c0, uint4 c1) {
    union { unsigned u[8]; v16bf b; } t;
    t.u[0] = c0.x; t.u[1] = c0.y; t.u[2] = c0.z; t.u[3] = c0.w;
    t.u[4] = c1.x; t.u[5] = c1.y; t.u[6] = c1.z; t.u[7] = c1.w;
    return t.b;
}

// A-matrix 16x32 bf16 fragment from LDS (row-major, stride Hh).
// ISA layout: lane half=0 holds K=[kb..kb+7] (v0..3) and K=[kb+16..kb+23] (v4..7);
//             lane half=1 holds K=[kb+8..+15] and K=[kb+24..+31]. Two b128 loads.
__device__ inline v16bf loadA_lds(const unsigned short* s, int mtile, int kbase,
                                  int half, int lr) {
    const unsigned short* row = s + (mtile * 16 + lr) * Hh + kbase;
    uint4 c0 = *reinterpret_cast<const uint4*>(row + half * 8);
    uint4 c1 = *reinterpret_cast<const uint4*>(row + 16 + half * 8);
    return pack16(c0, c1);
}
// B-matrix 32x16 bf16 fragment from global (weights stored transposed: Wt[n][k]).
// ISA layout: lanes 0-15 hold K=kb..kb+7 and kb+16..kb+23 for column n=lane; lanes 16-31 K=kb+8..+15 and kb+24..+31.
__device__ inline v16bf loadB_gl(const unsigned short* wt, int ntile, int kbase,
                                 int half, int lr) {
    const unsigned short* p = wt + (ntile * 16 + lr) * Hh + kbase + half * 8;
    uint4 c0 = *reinterpret_cast<const uint4*>(p);
    uint4 c1 = *reinterpret_cast<const uint4*>(p + 16);
    return pack16(c0, c1);
}

// Prep: bf16 + transpose weights into workspace.
//   ws[0     .. 16383] : Wt_enc[n][k]  = W_enc[k][n]
//   ws[16384 .. 32767] : Dt[0][n][k]   = f_W[0][k][n] - C_W[0][k][n]
//   ws[32768 .. 49151] : Dt[1][n][k]   = f_W[1][k][n] - C_W[1][k][n]
__global__ void commnet_prep(const float* __restrict__ W_enc,
                             const float* __restrict__ f_W,
                             const float* __restrict__ C_W,
                             unsigned short* __restrict__ ws) {
    int i = blockIdx.x * blockDim.x + threadIdx.x;
    if (i >= 3 * Hh * Hh) return;
    int mat = i >> 14, r = i & 16383;
    int n = r >> 7, k = r & 127;
    float v;
    if (mat == 0) v = W_enc[k * Hh + n];
    else          v = f_W[(mat - 1) * Hh * Hh + k * Hh + n]
                    - C_W[(mat - 1) * Hh * Hh + k * Hh + n];
    ws[mat * Hh * Hh + n * Hh + k] = f2bfbits(v);
}

__global__ __launch_bounds__(256) void commnet_main(
    const float* __restrict__ x,      const float* __restrict__ b_enc,
    const float* __restrict__ f_b,    const float* __restrict__ C_W,
    const float* __restrict__ C_b,    const float* __restrict__ head_W,
    const float* __restrict__ head_b, const float* __restrict__ val_W,
    const float* __restrict__ val_b,  const unsigned short* __restrict__ wt,
    float* __restrict__ out_action,   float* __restrict__ out_value) {
    __shared__ __align__(16) unsigned short sXe[MPAD * Hh];  // x_enc, bf16
    __shared__ __align__(16) unsigned short sH [MPAD * Hh];  // h, bf16
    __shared__ float sSum[Hh];
    __shared__ float sBias[Hh];

    const int tid  = threadIdx.x;
    const int wave = tid >> 5;      // 0..7 -> this wave's n-tile
    const int lane = tid & 31;
    const int half = lane >> 4;
    const int lr   = lane & 15;
    const int b    = blockIdx.x;    // batch

    // Stage x tile -> sH (bf16), zero-pad rows 100..111 (padded rows never feed
    // colsum or outputs, so their garbage evolution is harmless).
    for (int idx = tid; idx < MPAD * Hh; idx += 256) {
        int m = idx >> 7, k = idx & 127;
        float v = (m < Nn) ? x[((size_t)b * Nn + m) * Hh + k] : 0.f;
        sH[idx] = f2bfbits(v);
    }
    __syncthreads();

    const int n = wave * 16 + lr;

    // ---------------- Encoder: xe = tanh(x @ W_enc + b_enc) ----------------
    {
        v16bf bf0 = loadB_gl(wt, wave, 0,  half, lr);
        v16bf bf1 = loadB_gl(wt, wave, 32, half, lr);
        v16bf bf2 = loadB_gl(wt, wave, 64, half, lr);
        v16bf bf3 = loadB_gl(wt, wave, 96, half, lr);
        v8f acc[MT];
#pragma unroll
        for (int mt = 0; mt < MT; ++mt) {
            v8f c = {0.f, 0.f, 0.f, 0.f, 0.f, 0.f, 0.f, 0.f};
            c = __builtin_amdgcn_wmma_f32_16x16x32_bf16(false, loadA_lds(sH, mt, 0,  half, lr), false, bf0, (short)0, c, false, false);
            c = __builtin_amdgcn_wmma_f32_16x16x32_bf16(false, loadA_lds(sH, mt, 32, half, lr), false, bf1, (short)0, c, false, false);
            c = __builtin_amdgcn_wmma_f32_16x16x32_bf16(false, loadA_lds(sH, mt, 64, half, lr), false, bf2, (short)0, c, false, false);
            c = __builtin_amdgcn_wmma_f32_16x16x32_bf16(false, loadA_lds(sH, mt, 96, half, lr), false, bf3, (short)0, c, false, false);
            acc[mt] = c;
        }
        __syncthreads();                       // everyone done reading sH(x)
        const float bn = b_enc[n];
#pragma unroll
        for (int mt = 0; mt < MT; ++mt) {
#pragma unroll
            for (int r = 0; r < 8; ++r) {
                int m = mt * 16 + half * 8 + r;
                unsigned short hb = f2bfbits(tanhf(acc[mt][r] + bn));
                sXe[m * Hh + n] = hb;
                sH [m * Hh + n] = hb;          // h0 = x_enc
            }
        }
    }
    __syncthreads();

    // ------------- P layers: h = tanh(xe + h@(f_W-C_W) + (s@C_W + f_b + C_b)) -------------
    for (int layer = 0; layer < Pp; ++layer) {
        // colsum over the 100 real rows
        if (tid < Hh) {
            float s = 0.f;
            for (int m = 0; m < Nn; ++m) s += bf2f(sH[m * Hh + tid]);
            sSum[tid] = s;
        }
        __syncthreads();
        // per-batch GEMV: bias[n] = (s @ C_W[layer])[n] + f_b[n] + C_b[n]
        if (tid < Hh) {
            const float* cw = C_W + layer * Hh * Hh;
            float a = 0.f;
            for (int k = 0; k < Hh; ++k) a += sSum[k] * cw[k * Hh + tid];
            sBias[tid] = a + f_b[layer * Hh + tid] + C_b[layer * Hh + tid];
        }
        __syncthreads();

        const unsigned short* w = wt + (1 + layer) * Hh * Hh;   // Dt = f_W - C_W
        v16bf bf0 = loadB_gl(w, wave, 0,  half, lr);
        v16bf bf1 = loadB_gl(w, wave, 32, half, lr);
        v16bf bf2 = loadB_gl(w, wave, 64, half, lr);
        v16bf bf3 = loadB_gl(w, wave, 96, half, lr);
        v8f acc[MT];
#pragma unroll
        for (int mt = 0; mt < MT; ++mt) {
            v8f c = {0.f, 0.f, 0.f, 0.f, 0.f, 0.f, 0.f, 0.f};
            c = __builtin_amdgcn_wmma_f32_16x16x32_bf16(false, loadA_lds(sH, mt, 0,  half, lr), false, bf0, (short)0, c, false, false);
            c = __builtin_amdgcn_wmma_f32_16x16x32_bf16(false, loadA_lds(sH, mt, 32, half, lr), false, bf1, (short)0, c, false, false);
            c = __builtin_amdgcn_wmma_f32_16x16x32_bf16(false, loadA_lds(sH, mt, 64, half, lr), false, bf2, (short)0, c, false, false);
            c = __builtin_amdgcn_wmma_f32_16x16x32_bf16(false, loadA_lds(sH, mt, 96, half, lr), false, bf3, (short)0, c, false, false);
            acc[mt] = c;
        }
        __syncthreads();                       // everyone done reading old sH
        const float bias = sBias[n];
#pragma unroll
        for (int mt = 0; mt < MT; ++mt) {
#pragma unroll
            for (int r = 0; r < 8; ++r) {
                int m = mt * 16 + half * 8 + r;
                float v = tanhf(bf2f(sXe[m * Hh + n]) + acc[mt][r] + bias);
                sH[m * Hh + n] = f2bfbits(v);
            }
        }
        __syncthreads();
    }

    // ---------------- Heads: value GEMV + log-softmax over 5 logits ----------------
    for (int m = tid; m < Nn; m += 256) {
        float lg[Aa] = {0.f, 0.f, 0.f, 0.f, 0.f};
        float val = 0.f;
        for (int k = 0; k < Hh; ++k) {
            float hv = bf2f(sH[m * Hh + k]);
            val += hv * val_W[k];
#pragma unroll
            for (int a = 0; a < Aa; ++a) lg[a] += hv * head_W[k * Aa + a];
        }
        val += val_b[0];
#pragma unroll
        for (int a = 0; a < Aa; ++a) lg[a] += head_b[a];
        float mx = lg[0];
#pragma unroll
        for (int a = 1; a < Aa; ++a) mx = fmaxf(mx, lg[a]);
        float se = 0.f;
#pragma unroll
        for (int a = 0; a < Aa; ++a) se += expf(lg[a] - mx);
        float lse = mx + logf(se);
#pragma unroll
        for (int a = 0; a < Aa; ++a)
            out_action[((size_t)b * Nn + m) * Aa + a] = lg[a] - lse;
        out_value[(size_t)b * Nn + m] = val;
    }
}

extern "C" void kernel_launch(void* const* d_in, const int* in_sizes, int n_in,
                              void* d_out, int out_size, void* d_ws, size_t ws_size,
                              hipStream_t stream) {
    const float* x      = (const float*)d_in[0];
    const float* W_enc  = (const float*)d_in[1];
    const float* b_enc  = (const float*)d_in[2];
    const float* f_W    = (const float*)d_in[3];
    const float* f_b    = (const float*)d_in[4];
    const float* C_W    = (const float*)d_in[5];
    const float* C_b    = (const float*)d_in[6];
    const float* head_W = (const float*)d_in[7];
    const float* head_b = (const float*)d_in[8];
    const float* val_W  = (const float*)d_in[9];
    const float* val_b  = (const float*)d_in[10];
    // d_in[11] = comm_mask (ones - eye) folded analytically into colsum - self.

    unsigned short* wt = (unsigned short*)d_ws;    // 96 KB of bf16 weights
    float* out_action = (float*)d_out;             // [64,100,5]
    float* out_value  = out_action + (size_t)Bb * Nn * Aa;  // [64,100,1]

    commnet_prep<<<(3 * Hh * Hh + 255) / 256, 256, 0, stream>>>(W_enc, f_W, C_W, wt);
    commnet_main<<<Bb, 256, 0, stream>>>(x, b_enc, f_b, C_W, C_b, head_W, head_b,
                                         val_W, val_b, wt, out_action, out_value);
}
